// LSTMAutoencoder_70145405878280
// MI455X (gfx1250) — matrix operand order repaired and checked by reading, not compiled, not run on hardware
//
#include <hip/hip_runtime.h>

// ---------------------------------------------------------------------------
// Types for CDNA5 WMMA (wave32, v_wmma_f32_16x16x32_bf16)
// ---------------------------------------------------------------------------
typedef __attribute__((ext_vector_type(16))) __bf16 bf16x16;
typedef __attribute__((ext_vector_type(8)))  float  f32x8;
typedef __attribute__((ext_vector_type(4)))  unsigned u32x4;   // POD 128-bit

union FragAB {
    unsigned u[8];
    u32x4    q[2];
    bf16x16  v;
};

__device__ __forceinline__ unsigned short f2bf(float f) {
    union { float f; unsigned u; } c; c.f = f;
    unsigned u = c.u;
    u += 0x7FFFu + ((u >> 16) & 1u);   // round-to-nearest-even
    return (unsigned short)(u >> 16);
}

__device__ __forceinline__ unsigned pack_pair(float a, float b) {
    return (unsigned)f2bf(a) | ((unsigned)f2bf(b) << 16);
}

__device__ __forceinline__ float sigf(float x) { return 1.0f / (1.0f + __expf(-x)); }

__device__ __forceinline__ float tanh_f(float x) {
    x = fminf(fmaxf(x, -15.0f), 15.0f);
    float e = __expf(2.0f * x);
    return (e - 1.0f) / (e + 1.0f);
}

// ---------------------------------------------------------------------------
// A-fragment slot math (ISA 7.12.2 16-bit A layout), LDS pre-packed so a
// fragment load is 2 contiguous ds_load_b128 per lane.
// ---------------------------------------------------------------------------
__device__ __forceinline__ int a_slot(int m, int K) {
    int kt  = K >> 5;
    int kk  = K & 31;
    int grp = kk >> 4;
    int r   = kk & 15;
    int laneA = m + ((r >= 8) ? 16 : 0);
    int v   = grp * 4 + ((r & 7) >> 1);
    int par = K & 1;
    return ((kt * 32 + laneA) * 8 + v) * 2 + par;
}

__device__ __forceinline__ FragAB load_a_frag(const unsigned short* __restrict__ s,
                                              int kt, int lane) {
    FragAB f;
    const u32x4* p = (const u32x4*)(s + (size_t)(kt * 32 + lane) * 16);
    f.q[0] = p[0];
    f.q[1] = p[1];
    return f;
}

// B-fragment: weights pre-packed [tile][lane][8 dwords]; explicit global
// address space so these lower to global_load_b128 (LOADcnt only, no FLAT/DS
// coupling). `zo` is an opaque zero that defeats LICM hoist+spill across t.
typedef __attribute__((address_space(1))) const u32x4 g_u32x4;

__device__ __forceinline__ FragAB load_b_frag(const unsigned* __restrict__ wp,
                                              int tile_idx, int lane, int zo) {
    FragAB f;
    const g_u32x4* p =
        (const g_u32x4*)(wp + ((size_t)(tile_idx * 32 + lane + zo) << 3));
    f.q[0] = p[0];
    f.q[1] = p[1];
    return f;
}

__device__ __forceinline__ f32x8 wmma_bf16(const FragAB& a, const FragAB& b, f32x8 c) {
    return __builtin_amdgcn_wmma_f32_16x16x32_bf16(false, a.v, false, b.v,
                                                   (short)0, c, false, false);
}

// opaque zero: re-materialized every timestep so weight loads cannot be
// hoisted out of the 1024-step recurrence loop (which previously spilled
// 700+ VGPRs of fragments to scratch)
__device__ __forceinline__ int opaque_zero() {
    int z = 0;
    asm volatile("" : "+v"(z));
    return z;
}

// ---------------------------------------------------------------------------
// Problem constants
// ---------------------------------------------------------------------------
#define DIM_D   88
#define DIM_DP  96
#define DIM_H   256
#define DIM_4H  1024
#define DIM_L   128
#define NB      128
#define NT      1024
#define BTILE   16
#define KT_X    3
#define KT_H    8
#define NTIL_G  64
#define NTIL_O  6

#define N_WIH_P  (NTIL_G * KT_X * 256)   // 49152 u32
#define N_WHH_P  (NTIL_G * KT_H * 256)   // 131072 u32
#define N_WOUT_P (NTIL_O * KT_H * 256)   // 12288 u32

// ---------------------------------------------------------------------------
// Pack kernel: fp32 weights -> bf16 pair-packed B-fragment tiles
//   u32 index = ((nt*KT + kt)*32 + lane)*8 + v
//   n = nt*16 + (lane&15); K = kt*32 + (v>=4?16:0) + (lane<16?0:8) + 2(v&3)
// ---------------------------------------------------------------------------
__device__ __forceinline__ void pack_b_tiles(int i, const float* __restrict__ W,
                                             unsigned* __restrict__ out,
                                             int KTILES, int Kdim, int Ndim) {
    int gt   = i >> 8;
    int rem  = i & 255;
    int lane = rem >> 3;
    int v    = rem & 7;
    int nt   = gt / KTILES;
    int kt   = gt - nt * KTILES;
    int n    = nt * 16 + (lane & 15);
    int K    = kt * 32 + ((v >= 4) ? 16 : 0) + ((lane < 16) ? 0 : 8) + 2 * (v & 3);
    float w0 = (K     < Kdim && n < Ndim) ? W[n * Kdim + K]     : 0.0f;
    float w1 = (K + 1 < Kdim && n < Ndim) ? W[n * Kdim + K + 1] : 0.0f;
    out[i] = pack_pair(w0, w1);
}

__global__ __launch_bounds__(256)
void pack_kernel(const float* __restrict__ Wih_e, const float* __restrict__ Whh_e,
                 const float* __restrict__ bih_e, const float* __restrict__ bhh_e,
                 const float* __restrict__ Wih_d, const float* __restrict__ Whh_d,
                 const float* __restrict__ bih_d, const float* __restrict__ bhh_d,
                 const float* __restrict__ W_out,
                 unsigned* __restrict__ WihEP, unsigned* __restrict__ WhhEP,
                 unsigned* __restrict__ WihDP, unsigned* __restrict__ WhhDP,
                 unsigned* __restrict__ WoutP,
                 float* __restrict__ b_e, float* __restrict__ b_d) {
    int i = blockIdx.x * blockDim.x + threadIdx.x;
    if (i < N_WIH_P) { pack_b_tiles(i, Wih_e, WihEP, KT_X, DIM_D, DIM_4H); return; }
    i -= N_WIH_P;
    if (i < N_WHH_P) { pack_b_tiles(i, Whh_e, WhhEP, KT_H, DIM_H, DIM_4H); return; }
    i -= N_WHH_P;
    if (i < N_WIH_P) { pack_b_tiles(i, Wih_d, WihDP, KT_X, DIM_D, DIM_4H); return; }
    i -= N_WIH_P;
    if (i < N_WHH_P) { pack_b_tiles(i, Whh_d, WhhDP, KT_H, DIM_H, DIM_4H); return; }
    i -= N_WHH_P;
    if (i < N_WOUT_P) { pack_b_tiles(i, W_out, WoutP, KT_H, DIM_H, DIM_D); return; }
    i -= N_WOUT_P;
    if (i < DIM_4H) { b_e[i] = bih_e[i] + bhh_e[i]; return; }
    i -= DIM_4H;
    if (i < DIM_4H) { b_d[i] = bih_d[i] + bhh_d[i]; return; }
}

// ---------------------------------------------------------------------------
// Encoder
// ---------------------------------------------------------------------------
__global__ __launch_bounds__(256)
void enc_kernel(const float* __restrict__ x,          // [B,T,D]
                const unsigned* __restrict__ WihP,
                const unsigned* __restrict__ WhhP,
                const float* __restrict__ b_e,
                float* __restrict__ h_n) {            // [B,H]
    __shared__ __align__(16) unsigned short s_xa[KT_X * 32 * 16];  // 3 KB
    __shared__ __align__(16) unsigned short s_ha[KT_H * 32 * 16];  // 8 KB

    const int tid  = threadIdx.x;
    const int lane = tid & 31;
    const int wv   = tid >> 5;
    const int ln   = lane & 15;
    const int b0   = blockIdx.x * BTILE;
    const int m0   = (lane < 16) ? 0 : 8;

    for (int i = tid; i < KT_H * 32 * 8; i += 256) ((unsigned*)s_ha)[i] = 0u;

    int nt_idx[8];
#pragma unroll
    for (int g = 0; g < 4; ++g)
#pragma unroll
        for (int p = 0; p < 2; ++p) nt_idx[g * 2 + p] = 16 * g + 2 * wv + p;

    float bI[2], bF[2], bG[2], bO[2];
    int   slotH[16];
#pragma unroll
    for (int p = 0; p < 2; ++p) {
        const int n = 32 * wv + 16 * p + ln;
        bI[p] = b_e[n];
        bF[p] = b_e[256 + n];
        bG[p] = b_e[512 + n];
        bO[p] = b_e[768 + n];
#pragma unroll
        for (int r = 0; r < 8; ++r) slotH[p * 8 + r] = a_slot(m0 + r, n);
    }

    int    xslot[6];
    size_t xoff[6];
    int    xvalid[6];
#pragma unroll
    for (int j = 0; j < 6; ++j) {
        const int i = tid + j * 256;
        const int m = i / DIM_DP, d = i - m * DIM_DP;
        xslot[j]  = a_slot(m, d);
        xvalid[j] = (d < DIM_D);
        xoff[j]   = (size_t)(b0 + m) * (NT * DIM_D) + d;
    }

    float cst[16];
#pragma unroll
    for (int i = 0; i < 16; ++i) cst[i] = 0.0f;
    __syncthreads();

    for (int t = 0; t < NT; ++t) {
        const int zo = opaque_zero();

#pragma unroll
        for (int j = 0; j < 6; ++j) {
            float v = xvalid[j] ? x[xoff[j] + (size_t)t * DIM_D] : 0.0f;
            s_xa[xslot[j]] = f2bf(v);
        }
        __syncthreads();

        f32x8 acc[8];
#pragma unroll
        for (int i = 0; i < 8; ++i)
#pragma unroll
            for (int j = 0; j < 8; ++j) acc[i][j] = 0.0f;

#pragma unroll 1
        for (int kt = 0; kt < KT_X; ++kt) {
            FragAB a = load_a_frag(s_xa, kt, lane);
#pragma unroll
            for (int i = 0; i < 8; ++i) {
                FragAB b = load_b_frag(WihP, nt_idx[i] * KT_X + kt, lane, zo);
                acc[i] = wmma_bf16(a, b, acc[i]);
            }
        }
#pragma unroll 2
        for (int kt = 0; kt < KT_H; ++kt) {
            if (kt < KT_H - 1)
                __builtin_prefetch(
                    WhhP + (((size_t)(nt_idx[0] * KT_H + kt + 1) * 32 + lane) << 3), 0, 0);
            FragAB a = load_a_frag(s_ha, kt, lane);
#pragma unroll
            for (int i = 0; i < 8; ++i) {
                FragAB b = load_b_frag(WhhP, nt_idx[i] * KT_H + kt, lane, zo);
                acc[i] = wmma_bf16(a, b, acc[i]);
            }
        }
        __syncthreads();

#pragma unroll
        for (int p = 0; p < 2; ++p) {
#pragma unroll
            for (int r = 0; r < 8; ++r) {
                float gi = acc[0 + p][r] + bI[p];
                float gf = acc[2 + p][r] + bF[p];
                float gg = acc[4 + p][r] + bG[p];
                float go = acc[6 + p][r] + bO[p];
                float cc = sigf(gf) * cst[p * 8 + r] + sigf(gi) * tanh_f(gg);
                cst[p * 8 + r] = cc;
                float hv = sigf(go) * tanh_f(cc);
                s_ha[slotH[p * 8 + r]] = f2bf(hv);
                if (t == NT - 1)
                    h_n[(size_t)(b0 + m0 + r) * DIM_H + 32 * wv + 16 * p + ln] = hv;
            }
        }
        __syncthreads();
    }
}

// ---------------------------------------------------------------------------
// Latent bottleneck (tiny fp32 GEMVs)
// ---------------------------------------------------------------------------
__global__ __launch_bounds__(128)
void lat1_kernel(const float* __restrict__ h_n, const float* __restrict__ W_lat,
                 const float* __restrict__ b_lat, float* __restrict__ z_out) {
    const int b = blockIdx.x, l = threadIdx.x;
    const float* hr = h_n + b * DIM_H;
    const float* wr = W_lat + l * DIM_H;
    float s = b_lat[l];
    for (int k = 0; k < DIM_H; ++k) s += hr[k] * wr[k];
    z_out[b * DIM_L + l] = s;
}

__global__ __launch_bounds__(256)
void lat2_kernel(const float* __restrict__ z, const float* __restrict__ W_decinit,
                 const float* __restrict__ b_decinit, float* __restrict__ h0) {
    const int b = blockIdx.x, hu = threadIdx.x;
    const float* zr = z + b * DIM_L;
    const float* wr = W_decinit + hu * DIM_L;
    float s = b_decinit[hu];
    for (int l = 0; l < DIM_L; ++l) s += zr[l] * wr[l];
    h0[b * DIM_H + hu] = s;
}

// ---------------------------------------------------------------------------
// Decoder
// ---------------------------------------------------------------------------
__global__ __launch_bounds__(256)
void dec_kernel(const unsigned* __restrict__ WihP, const unsigned* __restrict__ WhhP,
                const unsigned* __restrict__ WoutP,
                const float* __restrict__ b_d, const float* __restrict__ b_out,
                const float* __restrict__ h0, float* __restrict__ out) {
    __shared__ __align__(16) unsigned short s_xa[KT_X * 32 * 16];
    __shared__ __align__(16) unsigned short s_ha[KT_H * 32 * 16];

    const int tid  = threadIdx.x;
    const int lane = tid & 31;
    const int wv   = tid >> 5;
    const int ln   = lane & 15;
    const int b0   = blockIdx.x * BTILE;
    const int m0   = (lane < 16) ? 0 : 8;

    for (int i = tid; i < KT_X * 32 * 8; i += 256) ((unsigned*)s_xa)[i] = 0u;
    for (int i = tid; i < BTILE * DIM_H; i += 256) {
        const int m = i >> 8, n = i & 255;
        s_ha[a_slot(m, n)] = f2bf(h0[(size_t)(b0 + m) * DIM_H + n]);
    }

    int nt_idx[8];
#pragma unroll
    for (int g = 0; g < 4; ++g)
#pragma unroll
        for (int p = 0; p < 2; ++p) nt_idx[g * 2 + p] = 16 * g + 2 * wv + p;

    float bI[2], bF[2], bG[2], bO[2];
    int   slotH[16];
#pragma unroll
    for (int p = 0; p < 2; ++p) {
        const int n = 32 * wv + 16 * p + ln;
        bI[p] = b_d[n];
        bF[p] = b_d[256 + n];
        bG[p] = b_d[512 + n];
        bO[p] = b_d[768 + n];
#pragma unroll
        for (int r = 0; r < 8; ++r) slotH[p * 8 + r] = a_slot(m0 + r, n);
    }

    const int n_o = wv * 16 + ln;
    float bo = 0.0f;
    int   slotX[8];
    if (wv < NTIL_O) {
        bo = (n_o < DIM_D) ? b_out[n_o] : 0.0f;
#pragma unroll
        for (int r = 0; r < 8; ++r) slotX[r] = a_slot(m0 + r, n_o);
    }

    float cst[16];
#pragma unroll
    for (int i = 0; i < 16; ++i) cst[i] = 0.0f;
    __syncthreads();

    for (int t = 0; t < NT; ++t) {
        const int zo = opaque_zero();

        f32x8 acc[8];
#pragma unroll
        for (int i = 0; i < 8; ++i)
#pragma unroll
            for (int j = 0; j < 8; ++j) acc[i][j] = 0.0f;

#pragma unroll 1
        for (int kt = 0; kt < KT_X; ++kt) {
            FragAB a = load_a_frag(s_xa, kt, lane);
#pragma unroll
            for (int i = 0; i < 8; ++i) {
                FragAB b = load_b_frag(WihP, nt_idx[i] * KT_X + kt, lane, zo);
                acc[i] = wmma_bf16(a, b, acc[i]);
            }
        }
#pragma unroll 2
        for (int kt = 0; kt < KT_H; ++kt) {
            if (kt < KT_H - 1)
                __builtin_prefetch(
                    WhhP + (((size_t)(nt_idx[0] * KT_H + kt + 1) * 32 + lane) << 3), 0, 0);
            FragAB a = load_a_frag(s_ha, kt, lane);
#pragma unroll
            for (int i = 0; i < 8; ++i) {
                FragAB b = load_b_frag(WhhP, nt_idx[i] * KT_H + kt, lane, zo);
                acc[i] = wmma_bf16(a, b, acc[i]);
            }
        }
        __syncthreads();

#pragma unroll
        for (int p = 0; p < 2; ++p) {
#pragma unroll
            for (int r = 0; r < 8; ++r) {
                float gi = acc[0 + p][r] + bI[p];
                float gf = acc[2 + p][r] + bF[p];
                float gg = acc[4 + p][r] + bG[p];
                float go = acc[6 + p][r] + bO[p];
                float cc = sigf(gf) * cst[p * 8 + r] + sigf(gi) * tanh_f(gg);
                cst[p * 8 + r] = cc;
                s_ha[slotH[p * 8 + r]] = f2bf(sigf(go) * tanh_f(cc));
            }
        }
        __syncthreads();

        if (wv < NTIL_O) {
            f32x8 pa;
#pragma unroll
            for (int j = 0; j < 8; ++j) pa[j] = 0.0f;
#pragma unroll 2
            for (int kt = 0; kt < KT_H; ++kt) {
                FragAB a = load_a_frag(s_ha, kt, lane);
                FragAB b = load_b_frag(WoutP, wv * KT_H + kt, lane, zo);
                pa = wmma_bf16(a, b, pa);
            }
#pragma unroll
            for (int r = 0; r < 8; ++r) {
                float pv = (n_o < DIM_D) ? sigf(pa[r] + bo) : 0.0f;
                s_xa[slotX[r]] = f2bf(pv);
                if (n_o < DIM_D)
                    out[((size_t)(b0 + m0 + r) * NT + t) * DIM_D + n_o] = pv;
            }
        }
        __syncthreads();
    }
}

// ---------------------------------------------------------------------------
// Host-side launcher
// ---------------------------------------------------------------------------
extern "C" void kernel_launch(void* const* d_in, const int* in_sizes, int n_in,
                              void* d_out, int out_size, void* d_ws, size_t ws_size,
                              hipStream_t stream) {
    const float* x         = (const float*)d_in[0];
    const float* Wih_e     = (const float*)d_in[1];
    const float* Whh_e     = (const float*)d_in[2];
    const float* bih_e     = (const float*)d_in[3];
    const float* bhh_e     = (const float*)d_in[4];
    const float* Wih_d     = (const float*)d_in[5];
    const float* Whh_d     = (const float*)d_in[6];
    const float* bih_d     = (const float*)d_in[7];
    const float* bhh_d     = (const float*)d_in[8];
    const float* W_lat     = (const float*)d_in[9];
    const float* b_lat     = (const float*)d_in[10];
    const float* W_decinit = (const float*)d_in[11];
    const float* b_decinit = (const float*)d_in[12];
    const float* W_out     = (const float*)d_in[13];
    const float* b_out     = (const float*)d_in[14];
    (void)in_sizes; (void)n_in; (void)out_size; (void)ws_size;

    float* out   = (float*)d_out;
    float* z_out = out + (size_t)NB * NT * DIM_D;   // tuple tail: z [B,L]

    unsigned* WihEP = (unsigned*)d_ws;
    unsigned* WhhEP = WihEP + N_WIH_P;
    unsigned* WihDP = WhhEP + N_WHH_P;
    unsigned* WhhDP = WihDP + N_WIH_P;
    unsigned* WoutP = WhhDP + N_WHH_P;
    float*    b_e   = (float*)(WoutP + N_WOUT_P);
    float*    b_d   = b_e + DIM_4H;
    float*    h_n   = b_d + DIM_4H;
    float*    h_d0  = h_n + NB * DIM_H;

    const int pack_total = 2 * (N_WIH_P + N_WHH_P) + N_WOUT_P + 2 * DIM_4H;
    pack_kernel<<<(pack_total + 255) / 256, 256, 0, stream>>>(
        Wih_e, Whh_e, bih_e, bhh_e, Wih_d, Whh_d, bih_d, bhh_d, W_out,
        WihEP, WhhEP, WihDP, WhhDP, WoutP, b_e, b_d);

    enc_kernel<<<NB / BTILE, 256, 0, stream>>>(x, WihEP, WhhEP, b_e, h_n);

    lat1_kernel<<<NB, DIM_L, 0, stream>>>(h_n, W_lat, b_lat, z_out);
    lat2_kernel<<<NB, DIM_H, 0, stream>>>(z_out, W_decinit, b_decinit, h_d0);

    dec_kernel<<<NB / BTILE, 256, 0, stream>>>(WihDP, WhhDP, WoutP, b_d, b_out,
                                               h_d0, out);
}
